// LFADS_32229434589317
// MI455X (gfx1250) — matrix operand order repaired and checked by reading, not compiled, not run on hardware
//
#include <hip/hip_runtime.h>

// ---------------------------------------------------------------------------
// LFADS forward pass for MI455X (gfx1250, wave32, WMMA bf16 16x16x32).
// B=512, T=256, N=E=128, F=32.
// All weights staged to LDS in WMMA B-fragment (swizzled) layout so that
// fragment loads are 2x ds_load_b128 per operand instead of 16x ds_load_u16.
// ---------------------------------------------------------------------------

typedef __attribute__((ext_vector_type(16))) __bf16 v16bf;
typedef __attribute__((ext_vector_type(8)))  float  v8f;

// ---- WMMA fragment helpers (wave32, 16x16x32 bf16 -> f32) -----------------
// A fragment: 16x32 bf16 tile, row-major, A points at (row0,k0), leading dim
// lda. Per lane: one contiguous 16-element (32B) run, k = half*16 + i.
__device__ __forceinline__ v16bf load_a16(const __bf16* A, int lda) {
  const int lane = threadIdx.x & 31;
  const int m = lane & 15;
  const int half = lane >> 4;
  v16bf f;
#pragma unroll
  for (int i = 0; i < 16; ++i) {
    int v = i >> 1, j = i & 1;
    int k = half * 16 + 2 * v + j;
    f[i] = A[m * lda + k];
  }
  return f;
}

// Swizzled B storage: a K x N weight matrix is tiled into (N/16)*(K/32)
// tiles of 32x16, tile order n-major then k. Each tile is 512 bf16 laid out
// so lane L's 16 elements are contiguous at tile + L*16:
//   lane = ((k>>4)&1)*16 + (n&15),  in-lane index = k&15.
template <int K, int N>
__device__ __forceinline__ int swz_idx(int k, int n) {
  constexpr int KT = K / 32;
  int tile = (n >> 4) * KT + (k >> 5);
  return tile * 512 + ((((k >> 4) & 1) << 4) + (n & 15)) * 16 + (k & 15);
}

// B fragment from swizzled tile: 16 contiguous bf16 per lane -> 2x ds_load_b128.
__device__ __forceinline__ v16bf load_b_swz(const __bf16* tile) {
  const __bf16* p = tile + (threadIdx.x & 31) * 16;
  v16bf f;
#pragma unroll
  for (int i = 0; i < 16; ++i) f[i] = p[i];
  return f;
}

__device__ __forceinline__ v8f wmma_bf16(v16bf a, v16bf b, v8f c) {
  return __builtin_amdgcn_wmma_f32_16x16x32_bf16(
      false, a, false, b, (short)0, c, false, false);
}

// C fragment store: lane = n + 16*(m/8), vgpr v = m%8.
__device__ __forceinline__ void store_c(float* C, int ldc, v8f acc) {
  const int lane = threadIdx.x & 31;
  const int n = lane & 15;
  const int g = lane >> 4;
#pragma unroll
  for (int v = 0; v < 8; ++v) C[(v + 8 * g) * ldc + n] = acc[v];
}

__device__ __forceinline__ float sigmoidf(float x) {
  return 1.0f / (1.0f + __expf(-x));
}

// ---------------------------------------------------------------------------
// K1: bidirectional GRU encoder. grid=(B/16, 2 dirs), 128 thr (4 waves).
// Each block owns 16 batch rows for all 256 steps; weights live in LDS (bf16).
// Per step: [x|h](16x256) @ Wzr(256x256) ; x @ Wxh(128x128) ; h @ Whh(128x128).
// ---------------------------------------------------------------------------
__global__ void __launch_bounds__(128)
lfads_enc_gru(const float* __restrict__ inputs,
              const float* __restrict__ fWx, const float* __restrict__ fWh,
              const float* __restrict__ fbi, const float* __restrict__ fbh,
              const float* __restrict__ bWx, const float* __restrict__ bWh,
              const float* __restrict__ bbi, const float* __restrict__ bbh,
              void* encoded_v) {
  constexpr int T = 256, N = 128;
  extern __shared__ char smem[];
  __bf16* sWzr  = (__bf16*)smem;               // 256x256 swz bf16 = 131072 B
  __bf16* sWxh  = (__bf16*)(smem + 131072);    // 128x128 swz bf16 =  32768 B
  __bf16* sWhh  = (__bf16*)(smem + 163840);    // 128x128 swz bf16 =  32768 B
  __bf16* sA    = (__bf16*)(smem + 196608);    // 16x256 [x|h]     =   8192 B
  float*  sPre  = (float*)(smem + 204800);     // 16x512 preact    =  32768 B
  float*  sBias = (float*)(smem + 237568);     // 512 f32          =   2048 B
  __bf16* encoded = (__bf16*)encoded_v;

  const int tid = threadIdx.x;
  const int dir = blockIdx.y;
  const int b0 = blockIdx.x * 16;
  const float* Wx = dir ? bWx : fWx;
  const float* Wh = dir ? bWh : fWh;
  const float* bi = dir ? bbi : fbi;
  const float* bh = dir ? bbh : fbh;

  // Stage weights (fp32 -> bf16, swizzled) once.
  for (int idx = tid; idx < 256 * 256; idx += 128) {
    int n = idx & 255, k = idx >> 8;
    float v = (k < 128) ? Wx[k * 384 + n] : Wh[(k - 128) * 384 + n];
    sWzr[swz_idx<256, 256>(k, n)] = (__bf16)v;
  }
  for (int idx = tid; idx < 128 * 128; idx += 128) {
    int n = idx & 127, k = idx >> 7;
    int d = swz_idx<128, 128>(k, n);
    sWxh[d] = (__bf16)Wx[k * 384 + 256 + n];
    sWhh[d] = (__bf16)Wh[k * 384 + 256 + n];
  }
  for (int idx = tid; idx < 512; idx += 128) {
    float v;
    if (idx < 256)      v = bi[idx] + bh[idx];   // z,r gates combined
    else if (idx < 384) v = bi[idx];             // xh bias (cols 256..383)
    else                v = bh[idx - 128];       // hh bias (cols 256..383)
    sBias[idx] = v;
  }
  for (int idx = tid; idx < 16 * 128; idx += 128)
    sA[(idx >> 7) * 256 + 128 + (idx & 127)] = (__bf16)0.0f;  // h = 0
  __syncthreads();

  const int wave = tid >> 5;
  const int lane = tid & 31;
  const int ncol = lane & 15;

  for (int s = 0; s < T; ++s) {
    const int t = dir ? (T - 1 - s) : s;
    for (int idx = tid; idx < 16 * 128; idx += 128) {
      int r = idx >> 7, c = idx & 127;
      sA[r * 256 + c] = (__bf16)inputs[((size_t)(b0 + r) * T + t) * N + c];
    }
    __syncthreads();

    // 32 output tiles (16 zr, 8 xh, 8 hh); 8 per wave.
    for (int tl = 0; tl < 8; ++tl) {
      int tile = wave * 8 + tl;
      v8f acc;
      if (tile < 16) {                  // zr: [x|h] @ Wzr, K=256 (KT=8)
        int nb = tile * 16;
        float bv = sBias[nb + ncol];
#pragma unroll
        for (int v = 0; v < 8; ++v) acc[v] = bv;
        for (int kc = 0; kc < 8; ++kc)
          acc = wmma_bf16(load_a16(sA + kc * 32, 256),
                          load_b_swz(sWzr + (tile * 8 + kc) * 512), acc);
        store_c(sPre + nb, 512, acc);
      } else if (tile < 24) {           // xh: x @ Wxh, K=128 (KT=4)
        int nt = tile - 16, nb = nt * 16;
        float bv = sBias[256 + nb + ncol];
#pragma unroll
        for (int v = 0; v < 8; ++v) acc[v] = bv;
        for (int kc = 0; kc < 4; ++kc)
          acc = wmma_bf16(load_a16(sA + kc * 32, 256),
                          load_b_swz(sWxh + (nt * 4 + kc) * 512), acc);
        store_c(sPre + 256 + nb, 512, acc);
      } else {                          // hh: h @ Whh, K=128 (KT=4)
        int nt = tile - 24, nb = nt * 16;
        float bv = sBias[384 + nb + ncol];
#pragma unroll
        for (int v = 0; v < 8; ++v) acc[v] = bv;
        for (int kc = 0; kc < 4; ++kc)
          acc = wmma_bf16(load_a16(sA + 128 + kc * 32, 256),
                          load_b_swz(sWhh + (nt * 4 + kc) * 512), acc);
        store_c(sPre + 384 + nb, 512, acc);
      }
    }
    __syncthreads();

    // GRU nonlinearity + state update + write encoded.
    for (int idx = tid; idx < 16 * 128; idx += 128) {
      int r = idx >> 7, c = idx & 127;
      const float* pr = sPre + r * 512;
      float z  = sigmoidf(pr[c]);
      float rr = sigmoidf(pr[128 + c]);
      float hc = tanhf(pr[256 + c] + rr * pr[384 + c]);
      float ho = (float)sA[r * 256 + 128 + c];
      float hn = z * ho + (1.0f - z) * hc;
      sA[r * 256 + 128 + c] = (__bf16)hn;
      encoded[(((size_t)(b0 + r) * T + t) * 256) + dir * 128 + c] = (__bf16)hn;
    }
    __syncthreads();
  }
}

// ---------------------------------------------------------------------------
// K2: reduced = flat(512x65536) @ denseW(65536x128), K split across grid.y,
// f32 atomic accumulation into pre-zeroed buffer. Prefetch next W chunk.
// ---------------------------------------------------------------------------
__global__ void __launch_bounds__(128)
lfads_dense(const void* __restrict__ encoded_v,
            const float* __restrict__ W, float* __restrict__ reduced) {
  extern __shared__ char smem[];
  __bf16* sB = (__bf16*)smem;  // 32x128 chunk of W, swizzled bf16 (8 tiles)
  const __bf16* enc = (const __bf16*)encoded_v;

  const int tid = threadIdx.x;
  const int wave = tid >> 5;
  const int lane = tid & 31;
  const int b0 = blockIdx.x * 16;
  const int k0 = blockIdx.y * 4096;
  const int nt0 = wave * 2;  // this wave owns col tiles nt0, nt0+1

  v8f acc0, acc1;
#pragma unroll
  for (int v = 0; v < 8; ++v) { acc0[v] = 0.0f; acc1[v] = 0.0f; }

  for (int kc = 0; kc < 128; ++kc) {
    const int kg = k0 + kc * 32;
    if (kc + 1 < 128)  // prefetch next 16KB W chunk (128B per thread)
      __builtin_prefetch((const char*)&W[(size_t)(kg + 32) * 128] + tid * 128,
                         0, 1);
    for (int idx = tid; idx < 32 * 128; idx += 128) {
      int n = idx & 127, k = idx >> 7;
      sB[swz_idx<32, 128>(k, n)] = (__bf16)W[(size_t)(kg + k) * 128 + n];
    }
    __syncthreads();
    v16bf a = load_a16(enc + (size_t)b0 * 65536 + kg, 65536);
    acc0 = wmma_bf16(a, load_b_swz(sB + nt0 * 512), acc0);
    acc1 = wmma_bf16(a, load_b_swz(sB + (nt0 + 1) * 512), acc1);
    __syncthreads();
  }
  const int n = lane & 15, g = lane >> 4;
#pragma unroll
  for (int v = 0; v < 8; ++v) {
    int row = b0 + v + 8 * g;
    atomicAdd(&reduced[row * 128 + nt0 * 16 + n], acc0[v]);
    atomicAdd(&reduced[row * 128 + nt0 * 16 + 16 + n], acc1[v]);
  }
}

// ---------------------------------------------------------------------------
// K3: mean = (reduced + dense_b) @ mean_W + mean_b; g0 = tanh(mean + e^.05*eps)
// Tiny (17 MFLOP): plain VALU, one block per batch row.
// ---------------------------------------------------------------------------
__global__ void __launch_bounds__(128)
lfads_latent(const float* __restrict__ reduced, const float* __restrict__ dense_b,
             const float* __restrict__ mean_W, const float* __restrict__ mean_b,
             const float* __restrict__ eps, void* g0_v) {
  extern __shared__ char smem[];
  float* sRow = (float*)smem;  // 128
  __bf16* g0 = (__bf16*)g0_v;
  const int b = blockIdx.x;
  const int e = threadIdx.x;
  sRow[e] = reduced[b * 128 + e] + dense_b[e];
  __syncthreads();
  float acc = mean_b[e];
  for (int k = 0; k < 128; ++k) acc = fmaf(sRow[k], mean_W[k * 128 + e], acc);
  float g = tanhf(acc + 1.0512710963760241f * eps[b * 128 + e]);  // exp(0.05)
  g0[b * 128 + e] = (__bf16)g;
}

// ---------------------------------------------------------------------------
// K4: generator GRU, 256 steps; grid = B/16 persistent blocks, weights in LDS.
// ru = sigmoid(h @ WgH + bg); c = tanh((r*h) @ WcH + bc); h = u*h + (1-u)*c.
// ---------------------------------------------------------------------------
__global__ void __launch_bounds__(128)
lfads_gen(const void* __restrict__ g0_v,
          const float* __restrict__ Wg, const float* __restrict__ bg,
          const float* __restrict__ Wc, const float* __restrict__ bc,
          void* gseq_v) {
  constexpr int T = 256;
  extern __shared__ char smem[];
  __bf16* sWg  = (__bf16*)smem;              // 128x256 swz = 65536 B
  __bf16* sWc  = (__bf16*)(smem + 65536);    // 128x128 swz = 32768 B
  __bf16* sH   = (__bf16*)(smem + 98304);    // 16x128  =  4096 B
  __bf16* sRH  = (__bf16*)(smem + 102400);   // 16x128  =  4096 B
  float*  sPre = (float*)(smem + 106496);    // 16x256  = 16384 B
  float*  sC   = (float*)(smem + 122880);    // 16x128  =  8192 B
  float*  sBg  = (float*)(smem + 131072);    // 256
  float*  sBc  = (float*)(smem + 132096);    // 128
  const __bf16* g0 = (const __bf16*)g0_v;
  __bf16* gseq = (__bf16*)gseq_v;

  const int tid = threadIdx.x;
  const int b0 = blockIdx.x * 16;
  for (int idx = tid; idx < 128 * 256; idx += 128) {
    int n = idx & 255, k = idx >> 8;
    sWg[swz_idx<128, 256>(k, n)] =
        (__bf16)Wg[(size_t)(128 + k) * 256 + n];
  }
  for (int idx = tid; idx < 128 * 128; idx += 128) {
    int n = idx & 127, k = idx >> 7;
    sWc[swz_idx<128, 128>(k, n)] =
        (__bf16)Wc[(size_t)(128 + k) * 128 + n];
  }
  for (int idx = tid; idx < 256; idx += 128) sBg[idx] = bg[idx];
  if (tid < 128) sBc[tid] = bc[tid];
  for (int idx = tid; idx < 16 * 128; idx += 128)
    sH[idx] = g0[(size_t)(b0 + (idx >> 7)) * 128 + (idx & 127)];
  __syncthreads();

  const int wave = tid >> 5;
  const int lane = tid & 31;
  const int ncol = lane & 15;

  for (int t = 0; t < T; ++t) {
    for (int tl = 0; tl < 4; ++tl) {  // ru: 16 tiles, 4/wave, K=128 (KT=4)
      int nt = wave * 4 + tl, nb = nt * 16;
      v8f acc;
      float bv = sBg[nb + ncol];
#pragma unroll
      for (int v = 0; v < 8; ++v) acc[v] = bv;
      for (int kc = 0; kc < 4; ++kc)
        acc = wmma_bf16(load_a16(sH + kc * 32, 128),
                        load_b_swz(sWg + (nt * 4 + kc) * 512), acc);
      store_c(sPre + nb, 256, acc);
    }
    __syncthreads();
    for (int idx = tid; idx < 16 * 128; idx += 128) {
      int r = idx >> 7, c = idx & 127;
      float rr = sigmoidf(sPre[r * 256 + c]);
      float u  = sigmoidf(sPre[r * 256 + 128 + c]);
      sPre[r * 256 + 128 + c] = u;  // stash u
      sRH[r * 128 + c] = (__bf16)(rr * (float)sH[r * 128 + c]);
    }
    __syncthreads();
    for (int tl = 0; tl < 2; ++tl) {  // c: 8 tiles, 2/wave, K=128 (KT=4)
      int nt = wave * 2 + tl, nb = nt * 16;
      v8f acc;
      float bv = sBc[nb + ncol];
#pragma unroll
      for (int v = 0; v < 8; ++v) acc[v] = bv;
      for (int kc = 0; kc < 4; ++kc)
        acc = wmma_bf16(load_a16(sRH + kc * 32, 128),
                        load_b_swz(sWc + (nt * 4 + kc) * 512), acc);
      store_c(sC + nb, 128, acc);
    }
    __syncthreads();
    for (int idx = tid; idx < 16 * 128; idx += 128) {
      int r = idx >> 7, c = idx & 127;
      float cc = tanhf(sC[r * 128 + c]);
      float u  = sPre[r * 256 + 128 + c];
      float ho = (float)sH[r * 128 + c];
      float hn = u * ho + (1.0f - u) * cc;
      sH[r * 128 + c] = (__bf16)hn;
      gseq[(((size_t)(b0 + r) * T + t) * 128) + c] = (__bf16)hn;
    }
    __syncthreads();
  }
}

// ---------------------------------------------------------------------------
// K5: fused factors+rates per 16-row tile of (B*T):
//   z = tanh(g @ facW + facb); out = log(dt) + 10*tanh(z @ neuW + neub).
// F=32 == one WMMA K-chunk for the second GEMM.
// ---------------------------------------------------------------------------
__global__ void __launch_bounds__(64)
lfads_rates(const void* __restrict__ gseq_v,
            const float* __restrict__ facW, const float* __restrict__ facb,
            const float* __restrict__ neuW, const float* __restrict__ neub,
            float* __restrict__ out) {
  extern __shared__ char smem[];
  __bf16* sFac = (__bf16*)smem;            // 128x32 swz (8 tiles)
  __bf16* sNeu = (__bf16*)(smem + 8192);   // 32x128 swz (8 tiles)
  __bf16* sZ   = (__bf16*)(smem + 16384);  // 16x32 row-major
  const __bf16* gseq = (const __bf16*)gseq_v;

  const int tid = threadIdx.x;
  const int wave = tid >> 5;
  const int lane = tid & 31;
  const int ncol = lane & 15;
  const int g = lane >> 4;
  const size_t row0 = (size_t)blockIdx.x * 16;

  for (int idx = tid; idx < 128 * 32; idx += 64) {
    int n = idx & 31, k = idx >> 5;
    sFac[swz_idx<128, 32>(k, n)] = (__bf16)facW[idx];
  }
  for (int idx = tid; idx < 32 * 128; idx += 64) {
    int n = idx & 127, k = idx >> 7;
    sNeu[swz_idx<32, 128>(k, n)] = (__bf16)neuW[idx];
  }
  __syncthreads();

  {  // z tile per wave (2 tiles of 16 cols), K=128 (KT=4)
    int nt = wave, nb = nt * 16;
    v8f acc;
    float bv = facb[nb + ncol];
#pragma unroll
    for (int v = 0; v < 8; ++v) acc[v] = bv;
    for (int kc = 0; kc < 4; ++kc)
      acc = wmma_bf16(load_a16(gseq + row0 * 128 + kc * 32, 128),
                      load_b_swz(sFac + (nt * 4 + kc) * 512), acc);
#pragma unroll
    for (int v = 0; v < 8; ++v)
      sZ[(v + 8 * g) * 32 + nb + ncol] = (__bf16)tanhf(acc[v]);
  }
  __syncthreads();

  for (int tl = 0; tl < 4; ++tl) {  // out: 8 tiles, 4/wave, K=32 (one chunk)
    int nt = wave * 4 + tl, nb = nt * 16;
    v8f acc;
    float bv = neub[nb + ncol];
#pragma unroll
    for (int v = 0; v < 8; ++v) acc[v] = bv;
    acc = wmma_bf16(load_a16(sZ, 32), load_b_swz(sNeu + nt * 512), acc);
#pragma unroll
    for (int v = 0; v < 8; ++v)
      out[(row0 + (size_t)(v + 8 * g)) * 128 + nb + ncol] =
          -4.605170185988091f + 10.0f * tanhf(acc[v]);  // log(0.01)
  }
}

// ---------------------------------------------------------------------------
extern "C" void kernel_launch(void* const* d_in, const int* in_sizes, int n_in,
                              void* d_out, int out_size, void* d_ws,
                              size_t ws_size, hipStream_t stream) {
  (void)in_sizes; (void)n_in; (void)out_size; (void)ws_size;
  const float* inputs = (const float*)d_in[0];
  const float* eps    = (const float*)d_in[1];
  const float* fWx = (const float*)d_in[2];
  const float* fWh = (const float*)d_in[3];
  const float* fbi = (const float*)d_in[4];
  const float* fbh = (const float*)d_in[5];
  const float* bWx = (const float*)d_in[6];
  const float* bWh = (const float*)d_in[7];
  const float* bbi = (const float*)d_in[8];
  const float* bbh = (const float*)d_in[9];
  const float* denseW = (const float*)d_in[10];
  const float* denseb = (const float*)d_in[11];
  const float* meanW  = (const float*)d_in[12];
  const float* meanb  = (const float*)d_in[13];
  const float* gateW  = (const float*)d_in[14];
  const float* gateb  = (const float*)d_in[15];
  const float* candW  = (const float*)d_in[16];
  const float* candb  = (const float*)d_in[17];
  const float* facW   = (const float*)d_in[18];
  const float* facb   = (const float*)d_in[19];
  const float* neuW   = (const float*)d_in[20];
  const float* neub   = (const float*)d_in[21];

  char* ws = (char*)d_ws;
  void*  encoded = ws;                               // B*T*2E bf16 = 64 MiB
  void*  gseq    = ws + 67108864ull;                 // B*T*E  bf16 = 32 MiB
  void*  g0buf   = ws + 100663296ull;                // B*E bf16
  float* reduced = (float*)(ws + 100794368ull);      // B*E f32

  hipMemsetAsync(reduced, 0, 512 * 128 * sizeof(float), stream);

  lfads_enc_gru<<<dim3(32, 2), 128, 239616, stream>>>(
      inputs, fWx, fWh, fbi, fbh, bWx, bWh, bbi, bbh, encoded);
  lfads_dense<<<dim3(32, 16), 128, 8192, stream>>>(encoded, denseW, reduced);
  lfads_latent<<<512, 128, 512, stream>>>(reduced, denseb, meanW, meanb, eps,
                                          g0buf);
  lfads_gen<<<32, 128, 132608, stream>>>(g0buf, gateW, gateb, candW, candb,
                                         gseq);
  lfads_rates<<<8192, 64, 17408, stream>>>(gseq, facW, facb, neuW, neub,
                                           (float*)d_out);
}
